// _PaiNNUpdateBlock_79663053406273
// MI455X (gfx1250) — matrix operand order repaired and checked by reading, not compiled, not run on hardware
//
#include <hip/hip_runtime.h>
#include <hip/hip_bf16.h>

// ---------------------------------------------------------------------------
// PaiNN update block, fused, bf16 WMMA (v_wmma_f32_16x16x32_bf16), gfx1250.
// ---------------------------------------------------------------------------

typedef __attribute__((ext_vector_type(16))) __bf16 v16bf;
typedef __attribute__((ext_vector_type(8)))  __bf16 v8bf;
typedef __attribute__((ext_vector_type(8)))  float  v8f;

union FragU { v16bf v; v8bf h[2]; };

// A-matrix 16x32 bf16 fragment from a row-major row (Krow >= koff+32 halfs).
// Contiguous k order (same as B): lane r in [0,16) holds M=r; elems 0..15 -> k = koff..koff+15
// (koff already includes hi*16).
__device__ __forceinline__ v16bf load_a_frag(const __bf16* rowBase, int koff) {
  FragU f;
  f.h[0] = *(const v8bf*)(rowBase + koff);
  f.h[1] = *(const v8bf*)(rowBase + koff + 8);
  return f.v;
}

// B-matrix 32x16 bf16 fragment from a per-lane column pointer (W[n][k] row-major
// over k). Contiguous k order (same as A): lane holds N=lane&15, elems 0..15 -> k = koff..koff+15
// (koff already includes hi*16).
__device__ __forceinline__ v16bf load_b_frag(const __bf16* colBase, int koff) {
  FragU f;
  f.h[0] = *(const v8bf*)(colBase + koff);
  f.h[1] = *(const v8bf*)(colBase + koff + 8);
  return f.v;
}

__device__ __forceinline__ v8f wmma_bf16(v16bf a, v16bf b, v8f c) {
  return __builtin_amdgcn_wmma_f32_16x16x32_bf16(
      /*neg_a=*/false, a, /*neg_b=*/false, b,
      /*c_mod=*/(short)0, c, /*reuse_a=*/false, /*reuse_b=*/false);
}

// ---------------------------------------------------------------------------
// Kernel 1: convert all weight matrices fp32 -> bf16 into workspace.
// Layout in ws (bf16 elements): [U_w 16384][V_w 16384][a_w1 32768][a_w2 49152]
// ---------------------------------------------------------------------------
__global__ __launch_bounds__(256) void convert_weights_kernel(
    const float* __restrict__ U, const float* __restrict__ V,
    const float* __restrict__ W1, const float* __restrict__ W2,
    __bf16* __restrict__ ws) {
  int i = blockIdx.x * 256 + threadIdx.x;
  if (i < 16384)        ws[i] = (__bf16)U[i];
  else if (i < 32768)   ws[i] = (__bf16)V[i - 16384];
  else if (i < 65536)   ws[i] = (__bf16)W1[i - 32768];
  else if (i < 114688)  ws[i] = (__bf16)W2[i - 65536];
}

// ---------------------------------------------------------------------------
// Kernel 2: fused PaiNN update. One workgroup (8 wave32) per 16-atom tile.
// Wave w owns output-feature columns [16w, 16w+16).
// ---------------------------------------------------------------------------
__global__ __launch_bounds__(256) void painn_update_kernel(
    const float* __restrict__ s,  const float* __restrict__ v,
    const __bf16* __restrict__ wU,  const __bf16* __restrict__ wV,
    const __bf16* __restrict__ wW1, const __bf16* __restrict__ wW2,
    const float* __restrict__ b1, const float* __restrict__ b2,
    float* __restrict__ out_s, float* __restrict__ out_v, int N) {

  __shared__ __align__(16) __bf16 vA[3][16][128];   // v tile, per channel   (12KB)
  __shared__ __align__(16) __bf16 mlpA[16][256];    // [s | ||Vv||^2] tile   ( 8KB)
  __shared__ __align__(16) __bf16 hA[16][128];      // softplus hidden tile  ( 4KB)

  const int tid  = threadIdx.x;
  const int wv   = tid >> 5;
  const int lane = tid & 31;
  const int r    = lane & 15;
  const int hi   = lane >> 4;
  const int m0   = blockIdx.x * 16;

  // Speculative prefetch of next tile's inputs (global_prefetch).
  {
    const int nb = (blockIdx.x + 1) * 16;
    if (nb * 384 + 6143 < N * 384) {
      if (tid < 192) __builtin_prefetch(v + (size_t)nb * 384 + tid * 32, 0, 0);
      else           __builtin_prefetch(s + (size_t)nb * 128 + (tid - 192) * 32, 0, 0);
    }
  }

  // ---- Stage 0: v tile -> vA[c][m][f] bf16 ; s tile -> mlpA[m][0..127] ----
  for (int i = tid; i < 16 * 384; i += 256) {
    int m = i / 384;
    int rem = i - m * 384;
    int f = rem / 3;
    int c = rem - f * 3;
    int mm = m0 + m; if (mm >= N) mm = N - 1;
    vA[c][m][f] = (__bf16)v[(size_t)mm * 384 + rem];
  }
  for (int i = tid; i < 16 * 128; i += 256) {
    int m = i >> 7;
    int f = i & 127;
    int mm = m0 + m; if (mm >= N) mm = N - 1;
    mlpA[m][f] = (__bf16)s[(size_t)mm * 128 + f];
  }
  __syncthreads();

  const int g = wv * 16 + r;   // this lane's output feature column

  // ---- GEMM 1: Uv, Vv  (K=128, 3 channels) ----
  v8f accU[3], accV[3];
  #pragma unroll
  for (int c = 0; c < 3; ++c) { accU[c] = {}; accV[c] = {}; }

  #pragma unroll
  for (int c = 0; c < 3; ++c) {
    const __bf16* arow = &vA[c][r][0];
    const __bf16* ucol = wU + (size_t)g * 128;
    const __bf16* vcol = wV + (size_t)g * 128;
    #pragma unroll
    for (int kt = 0; kt < 4; ++kt) {
      v16bf a  = load_a_frag(arow, kt * 32 + hi * 16);
      v16bf bu = load_b_frag(ucol, kt * 32 + hi * 16);
      v16bf bv = load_b_frag(vcol, kt * 32 + hi * 16);
      accU[c] = wmma_bf16(a, bu, accU[c]);
      accV[c] = wmma_bf16(a, bv, accV[c]);
    }
  }

  // ---- inner = <Uv,Vv>, ||Vv||^2 -> mlpA[:,128+g] ----
  float inner[8];
  #pragma unroll
  for (int i = 0; i < 8; ++i) {
    float u0 = accU[0][i], u1 = accU[1][i], u2 = accU[2][i];
    float w0 = accV[0][i], w1 = accV[1][i], w2 = accV[2][i];
    inner[i] = u0 * w0 + u1 * w1 + u2 * w2;
    float nrm = w0 * w0 + w1 * w1 + w2 * w2;
    mlpA[i + hi * 8][128 + g] = (__bf16)nrm;
  }
  __syncthreads();

  // ---- GEMM 2: h = ssp(mlp_in @ a_w1^T + b1)  (K=256) ----
  v8f acch = {};
  {
    const __bf16* arow = &mlpA[r][0];
    const __bf16* bcol = wW1 + (size_t)g * 256;
    #pragma unroll
    for (int kt = 0; kt < 8; ++kt) {
      v16bf a = load_a_frag(arow, kt * 32 + hi * 16);
      v16bf b = load_b_frag(bcol, kt * 32 + hi * 16);
      acch = wmma_bf16(a, b, acch);
    }
  }
  const float bias1 = b1[g];
  #pragma unroll
  for (int i = 0; i < 8; ++i) {
    float x = acch[i] + bias1;
    // shifted softplus: log(1+exp(x)) - log(2), numerically stable form
    float t  = fabsf(x);
    float sp = fmaxf(x, 0.0f) + __logf(1.0f + __expf(-t)) - 0.69314718056f;
    hA[i + hi * 8][g] = (__bf16)sp;
  }
  __syncthreads();

  // ---- GEMM 3: a = h @ a_w2^T + b2  (K=128, three 16-col output tiles that
  //      all land on feature column g: a_ss / a_sv / a_vv) ----
  v8f accA[3];
  #pragma unroll
  for (int t = 0; t < 3; ++t) accA[t] = {};
  {
    const __bf16* arow = &hA[r][0];
    #pragma unroll
    for (int t = 0; t < 3; ++t) {
      const __bf16* bcol = wW2 + (size_t)(t * 128 + g) * 128;
      #pragma unroll
      for (int kt = 0; kt < 4; ++kt) {
        v16bf a = load_a_frag(arow, kt * 32 + hi * 16);
        v16bf b = load_b_frag(bcol, kt * 32 + hi * 16);
        accA[t] = wmma_bf16(a, b, accA[t]);
      }
    }
  }

  // ---- Fused epilogue: s_new, v_new ----
  const float bss = b2[g], bsv = b2[128 + g], bvv = b2[256 + g];
  #pragma unroll
  for (int i = 0; i < 8; ++i) {
    int m = m0 + i + hi * 8;
    if (m >= N) continue;
    float a_ss = accA[0][i] + bss;
    float a_sv = accA[1][i] + bsv;
    float a_vv = accA[2][i] + bvv;

    size_t so = (size_t)m * 128 + g;
    out_s[so] = s[so] + a_ss + a_sv * inner[i];

    size_t vo = (size_t)m * 384 + (size_t)g * 3;
    out_v[vo + 0] = v[vo + 0] + a_vv * accU[0][i];
    out_v[vo + 1] = v[vo + 1] + a_vv * accU[1][i];
    out_v[vo + 2] = v[vo + 2] + a_vv * accU[2][i];
  }
}

// ---------------------------------------------------------------------------
extern "C" void kernel_launch(void* const* d_in, const int* in_sizes, int n_in,
                              void* d_out, int out_size, void* d_ws, size_t ws_size,
                              hipStream_t stream) {
  const float* s    = (const float*)d_in[0];
  const float* v    = (const float*)d_in[1];
  const float* U_w  = (const float*)d_in[2];
  const float* V_w  = (const float*)d_in[3];
  const float* a_w1 = (const float*)d_in[4];
  const float* a_b1 = (const float*)d_in[5];
  const float* a_w2 = (const float*)d_in[6];
  const float* a_b2 = (const float*)d_in[7];

  const int F = 128;
  const int N = in_sizes[0] / F;   // s is (N, F)

  __bf16* ws  = (__bf16*)d_ws;
  __bf16* wU  = ws;             // 16384 elems
  __bf16* wV  = ws + 16384;     // 16384 elems
  __bf16* wW1 = ws + 32768;     // 32768 elems (F x 2F)
  __bf16* wW2 = ws + 65536;     // 49152 elems (3F x F)

  const int totalW = 16384 + 16384 + 32768 + 49152;   // 114688 elems (224KB)
  convert_weights_kernel<<<(totalW + 255) / 256, 256, 0, stream>>>(
      U_w, V_w, a_w1, a_w2, ws);

  float* out_s = (float*)d_out;
  float* out_v = out_s + (size_t)N * F;

  const int blocks = (N + 15) / 16;   // N=100000 -> 6250
  painn_update_kernel<<<blocks, 256, 0, stream>>>(
      s, v, wU, wV, wW1, wW2, a_b1, a_b2, out_s, out_v, N);
}
